// DeepFM_7808250544784
// MI455X (gfx1250) — matrix operand order repaired and checked by reading, not compiled, not run on hardware
//
#include <hip/hip_runtime.h>
#include <hip/hip_bf16.h>

// ---------------- problem constants ----------------
#define BATCH     16384
#define NFEAT     26
#define VOCAB     100000
#define EDIM      16
#define NINT      13
#define DEEP_REAL 429          // NFEAT*EDIM + NINT
#define DEEP_K    448          // padded to multiple of 32 (14 K-tiles)

typedef __bf16 bf16_t;
typedef bf16_t bf16x16 __attribute__((ext_vector_type(16)));
typedef bf16_t bf16x8  __attribute__((ext_vector_type(8)));
typedef float  f32x8   __attribute__((ext_vector_type(8)));
typedef int    i32x4   __attribute__((vector_size(4 * sizeof(int))));

#define AS1 __attribute__((address_space(1)))
#define AS3 __attribute__((address_space(3)))

#if defined(__has_builtin)
#if __has_builtin(__builtin_amdgcn_global_load_async_to_lds_b128)
#define HAVE_ASYNC_LDS 1
#endif
#endif
#ifndef HAVE_ASYNC_LDS
#define HAVE_ASYNC_LDS 0
#endif

__device__ __forceinline__ bf16_t f2bf(float f) {
    __hip_bfloat16 h = __float2bfloat16(f);
    return *reinterpret_cast<bf16_t*>(&h);
}
__device__ __forceinline__ float bf2f(bf16_t x) {
    __hip_bfloat16 h = *reinterpret_cast<__hip_bfloat16*>(&x);
    return __bfloat162float(h);
}

// ---------------------------------------------------------------------------
// Pack a row-major fp32 weight [Kreal x N] into bf16 WMMA B-fragment order,
// zero-padding K up to Kpad.  Layout: [Kpad/32][N/16] tiles of 512 bf16,
// within a tile: lane-major (32 lanes x 16 elems).
//   lane L (0..31): n = nt*16 + (L&15), k = kt*32 + (L<16 ? 0 : 8) + (e&7) + (e>=8 ? 16 : 0)
// ---------------------------------------------------------------------------
__global__ __launch_bounds__(256) void pack_weight_kernel(
    const float* __restrict__ W, bf16_t* __restrict__ Wp,
    int Kreal, int Kpad, int N)
{
    const int idx = blockIdx.x * 256 + threadIdx.x;
    const int total = Kpad * N;
    if (idx >= total) return;
    const int e    = idx & 15;
    const int lane = (idx >> 4) & 31;
    const int tile = idx >> 9;
    const int ntN  = N >> 4;
    const int kt   = tile / ntN;
    const int nt   = tile - kt * ntN;
    const int n    = (nt << 4) + (lane & 15);
    const int k    = (kt << 5) + ((lane < 16) ? 0 : 8) + (e & 7) + ((e >> 3) << 4);
    const float v  = (k < Kreal) ? W[(size_t)k * N + n] : 0.0f;
    Wp[idx] = f2bf(v);
}

// ---------------------------------------------------------------------------
// Front: embedding gather + FM 2nd-order + linear terms + deep_in (bf16).
// One wave32 per batch row; 8 rows per 256-thread block.
// ---------------------------------------------------------------------------
__global__ __launch_bounds__(256) void front_kernel(
    const float* __restrict__ x_int, const int* __restrict__ x_cat,
    const float* __restrict__ emb,   const float* __restrict__ lin,
    const float* __restrict__ W_num, const float* __restrict__ b_num,
    const float* __restrict__ bias0,
    bf16_t* __restrict__ deep_in, float* __restrict__ partial)
{
    const int lane = threadIdx.x & 31;
    const int wave = threadIdx.x >> 5;
    const int b    = blockIdx.x * 8 + wave;
    if (b >= BATCH) return;

    const int* row_cat = x_cat + (size_t)b * NFEAT;
    bf16_t*    drow    = deep_in + (size_t)b * DEEP_K;

    float fmp = 0.0f, ls = 0.0f, ns = 0.0f;

    if (lane < EDIM) {
        const int d = lane;
        float s = 0.0f, ss = 0.0f;
        #pragma unroll
        for (int f = 0; f < NFEAT; ++f) {
            const int   idx = row_cat[f];
            const float e   = emb[((size_t)f * VOCAB + (size_t)idx) * EDIM + d];
            s  += e;
            ss += e * e;
            drow[f * EDIM + d] = f2bf(e);
        }
        fmp = s * s - ss;                      // contribution of dim d
        fmp += __shfl_xor(fmp, 8, 32);         // reduce across lanes 0..15
        fmp += __shfl_xor(fmp, 4, 32);
        fmp += __shfl_xor(fmp, 2, 32);
        fmp += __shfl_xor(fmp, 1, 32);
    } else if (lane == 16) {
        for (int f = 0; f < NFEAT; ++f)
            ls += lin[(size_t)f * VOCAB + (size_t)row_cat[f]];
    } else if (lane == 17) {
        ns = b_num[0];
        for (int j = 0; j < NINT; ++j) {
            const float x = x_int[(size_t)b * NINT + j];
            ns += x * W_num[j];
            drow[NFEAT * EDIM + j] = f2bf(x);
        }
    } else if (lane == 18) {
        for (int c = DEEP_REAL; c < DEEP_K; ++c) drow[c] = f2bf(0.0f);
    }

    const float ls_all = __shfl(ls, 16, 32);
    const float ns_all = __shfl(ns, 17, 32);
    const float fm_all = __shfl(fmp, 0, 32);
    if (lane == 0)
        partial[b] = bias0[0] + ls_all + ns_all + 0.5f * fm_all;
}

// ---------------------------------------------------------------------------
// bf16 WMMA GEMM:  C[M x N] = relu?( A[M x K] @ Bpacked + bias )
// 256 threads = 8 waves; block tile = 256 rows x 64 cols.
// The 64-col B panel (kTiles x 4 tiles x 1KB) is staged once into LDS with
// CDNA5 async-to-LDS copies (ASYNCcnt + s_wait_asynccnt), then all 8 waves
// feed WMMA from ds_load.  Each wave owns TWO 16-row M-tiles x 4 N-tiles
// (8 accumulators): per K-step 4 global A loads + 4 LDS B reads -> 8 WMMAs.
// A: row-major bf16.  Fragment per ISA 16-bit A layout:
//   lanes 0..15:  elems 0..7 -> K+0..7,  elems 8..15 -> K+16..23
//   lanes 16..31: elems 0..7 -> K+8..15, elems 8..15 -> K+24..31
// ---------------------------------------------------------------------------
__global__ __launch_bounds__(256) void gemm_bf16_wmma(
    const bf16_t* __restrict__ A, const bf16_t* __restrict__ Bp,
    const float* __restrict__ bias, bf16_t* __restrict__ C,
    int M, int N, int K, int relu)
{
    extern __shared__ char smem_raw[];       // kTiles * 4096 bytes
    const int tid  = threadIdx.x;
    const int lane = tid & 31;
    const int wave = tid >> 5;
    const int m0   = blockIdx.x * 256 + wave * 32;
    const int n0   = blockIdx.y * 64;
    const int half = lane >> 4;              // 0 or 1
    const int l16  = lane & 15;
    const int ntN  = N >> 4;
    const int kTiles = K >> 5;

    // ---- stage the block's B panel into LDS (one 16B chunk/thread/kt) ----
    for (int kt = 0; kt < kTiles; ++kt) {
        const bf16_t* src = Bp + (((size_t)kt * ntN + (n0 >> 4)) << 9) + tid * 8;
        char*         dst = smem_raw + kt * 4096 + tid * 16;
#if HAVE_ASYNC_LDS
        __builtin_amdgcn_global_load_async_to_lds_b128(
            (AS1 i32x4*)src, (AS3 i32x4*)dst, 0, 0);
#else
        *(i32x4*)dst = *(const i32x4*)src;
#endif
    }
#if HAVE_ASYNC_LDS
    asm volatile("s_wait_asynccnt 0x0" ::: "memory");
#endif
    __syncthreads();

    f32x8 acc0[4] = {};
    f32x8 acc1[4] = {};

    const bf16_t* arow0 = A + (size_t)(m0 + l16) * K + half * 8;
    const bf16_t* arow1 = arow0 + (size_t)16 * K;

    for (int kt = 0; kt < kTiles; ++kt) {
        bf16x16 a0, a1;
        {
            const bf16_t* ap = arow0 + (kt << 5);
            bf16x8 lo = *reinterpret_cast<const bf16x8*>(ap);
            bf16x8 hi = *reinterpret_cast<const bf16x8*>(ap + 16);
            #pragma unroll
            for (int i = 0; i < 8; ++i) { a0[i] = lo[i]; a0[8 + i] = hi[i]; }
        }
        {
            const bf16_t* ap = arow1 + (kt << 5);
            bf16x8 lo = *reinterpret_cast<const bf16x8*>(ap);
            bf16x8 hi = *reinterpret_cast<const bf16x8*>(ap + 16);
            #pragma unroll
            for (int i = 0; i < 8; ++i) { a1[i] = lo[i]; a1[8 + i] = hi[i]; }
        }
        const bf16_t* bb = (const bf16_t*)(smem_raw + kt * 4096) + (lane << 4);
        #pragma unroll
        for (int j = 0; j < 4; ++j) {
            bf16x16 b = *reinterpret_cast<const bf16x16*>(bb + (j << 9));
            acc0[j] = __builtin_amdgcn_wmma_f32_16x16x32_bf16(
                false, a0, false, b, (short)0, acc0[j], false, false);
            acc1[j] = __builtin_amdgcn_wmma_f32_16x16x32_bf16(
                false, a1, false, b, (short)0, acc1[j], false, false);
        }
    }

    // Epilogue: C/D layout -> VGPR r holds row base + half*8 + r, col = n0+j*16+l16
    const int rb0 = m0 + half * 8;
    const int rb1 = rb0 + 16;
    #pragma unroll
    for (int j = 0; j < 4; ++j) {
        const int   col = n0 + j * 16 + l16;
        const float bs  = bias[col];
        #pragma unroll
        for (int r = 0; r < 8; ++r) {
            float v0 = acc0[j][r] + bs;
            float v1 = acc1[j][r] + bs;
            if (relu) { v0 = fmaxf(v0, 0.0f); v1 = fmaxf(v1, 0.0f); }
            C[(size_t)(rb0 + r) * N + col] = f2bf(v0);
            C[(size_t)(rb1 + r) * N + col] = f2bf(v1);
        }
    }
}

// ---------------------------------------------------------------------------
// Final: out[b] = partial[b] + b5 + h4[b, 0:64] . W5   (fp32)
// ---------------------------------------------------------------------------
__global__ __launch_bounds__(256) void final_kernel(
    const bf16_t* __restrict__ h4, const float* __restrict__ W5,
    const float* __restrict__ b5, const float* __restrict__ partial,
    float* __restrict__ out)
{
    const int b = blockIdx.x * 256 + threadIdx.x;
    if (b >= BATCH) return;
    const bf16_t* hr = h4 + (size_t)b * 64;
    float s = 0.0f;
    #pragma unroll
    for (int i = 0; i < 64; ++i)
        s += bf2f(hr[i]) * W5[i];
    out[b] = partial[b] + b5[0] + s;
}

// ---------------------------------------------------------------------------
extern "C" void kernel_launch(void* const* d_in, const int* in_sizes, int n_in,
                              void* d_out, int out_size, void* d_ws, size_t ws_size,
                              hipStream_t stream) {
    const float* x_int = (const float*)d_in[0];
    const int*   x_cat = (const int*)  d_in[1];
    const float* emb   = (const float*)d_in[2];
    const float* lin   = (const float*)d_in[3];
    const float* W_num = (const float*)d_in[4];
    const float* b_num = (const float*)d_in[5];
    const float* bias0 = (const float*)d_in[6];
    const float* W1 = (const float*)d_in[7];  const float* b1 = (const float*)d_in[8];
    const float* W2 = (const float*)d_in[9];  const float* b2 = (const float*)d_in[10];
    const float* W3 = (const float*)d_in[11]; const float* b3 = (const float*)d_in[12];
    const float* W4 = (const float*)d_in[13]; const float* b4 = (const float*)d_in[14];
    const float* W5 = (const float*)d_in[15]; const float* b5 = (const float*)d_in[16];
    float* out = (float*)d_out;

    // workspace carve-up (256B aligned)
    char*  ws  = (char*)d_ws;
    size_t off = 0;
    auto carve = [&](size_t bytes) -> char* {
        char* p = ws + off;
        off = (off + bytes + 255) & ~(size_t)255;
        return p;
    };
    float*  partial = (float*) carve((size_t)BATCH * 4);
    bf16_t* deep_in = (bf16_t*)carve((size_t)BATCH * DEEP_K * 2);
    bf16_t* h1      = (bf16_t*)carve((size_t)BATCH * 512 * 2);
    bf16_t* h2      = (bf16_t*)carve((size_t)BATCH * 256 * 2);
    bf16_t* h3      = (bf16_t*)carve((size_t)BATCH * 128 * 2);
    bf16_t* h4      = (bf16_t*)carve((size_t)BATCH * 64 * 2);
    bf16_t* W1p     = (bf16_t*)carve((size_t)DEEP_K * 512 * 2);
    bf16_t* W2p     = (bf16_t*)carve((size_t)512 * 256 * 2);
    bf16_t* W3p     = (bf16_t*)carve((size_t)256 * 128 * 2);
    bf16_t* W4p     = (bf16_t*)carve((size_t)128 * 64 * 2);

    // 1) pack weights into WMMA B-fragment order (bf16, K padded)
    auto packGrid = [](int kpad, int n) { return dim3((kpad * n + 255) / 256); };
    pack_weight_kernel<<<packGrid(DEEP_K, 512), 256, 0, stream>>>(W1, W1p, DEEP_REAL, DEEP_K, 512);
    pack_weight_kernel<<<packGrid(512, 256),   256, 0, stream>>>(W2, W2p, 512, 512, 256);
    pack_weight_kernel<<<packGrid(256, 128),   256, 0, stream>>>(W3, W3p, 256, 256, 128);
    pack_weight_kernel<<<packGrid(128, 64),    256, 0, stream>>>(W4, W4p, 128, 128, 64);

    // 2) embedding gather + FM + linear + deep_in
    front_kernel<<<dim3(BATCH / 8), 256, 0, stream>>>(
        x_int, x_cat, emb, lin, W_num, b_num, bias0, deep_in, partial);

    // 3) MLP via bf16 WMMA (block tile 256x64, B panel staged in LDS)
    const int kt1 = DEEP_K / 32, kt2 = 512 / 32, kt3 = 256 / 32, kt4 = 128 / 32;
    gemm_bf16_wmma<<<dim3(BATCH / 256, 512 / 64), 256, kt1 * 4096, stream>>>(
        deep_in, W1p, b1, h1, BATCH, 512, DEEP_K, 1);
    gemm_bf16_wmma<<<dim3(BATCH / 256, 256 / 64), 256, kt2 * 4096, stream>>>(
        h1, W2p, b2, h2, BATCH, 256, 512, 1);
    gemm_bf16_wmma<<<dim3(BATCH / 256, 128 / 64), 256, kt3 * 4096, stream>>>(
        h2, W3p, b3, h3, BATCH, 128, 256, 1);
    gemm_bf16_wmma<<<dim3(BATCH / 256, 64 / 64), 256, kt4 * 4096, stream>>>(
        h3, W4p, b4, h4, BATCH, 64, 128, 1);

    // 4) final score
    final_kernel<<<dim3(BATCH / 256), 256, 0, stream>>>(h4, W5, b5, partial, out);
}
